// SpConAtt_72859825209521
// MI455X (gfx1250) — compile-verified, matrix operand order checked by reading, not run on hardware
//
#include <hip/hip_runtime.h>
#include <hip/hip_bf16.h>
#include <math.h>

// ---------------------------------------------------------------------------
// SpConAtt on MI455X (gfx1250, wave32):
//   K1 : QKV 1x1 projection (+positional channels) -> f16 Q/K, f32 V
//   K1b: repack Wout -> Wb[kl][o][c] f16 (WMMA B-operand friendly)
//   K2 : softmax denominators via full Q.K^T row sums, v_wmma_f32_16x16x32_f16,
//        4 waves/block sharing LDS double-buffered K tiles (software pipelined)
//   K3 : local 7x7 window attention * V contracted with Wout via WMMA
// Working set ~13MB -> fully L2-resident (192MB); matrix-core / LDS bound.
// ---------------------------------------------------------------------------

typedef __attribute__((ext_vector_type(16))) _Float16 v16h;
typedef __attribute__((ext_vector_type(8)))  _Float16 v8h;
typedef __attribute__((ext_vector_type(8)))  float    v8f;

#define B_   2
#define CIN  256
#define H_   64
#define W_   64
#define HW   4096
#define C_   64
#define QC   60
#define KL   49
#define R_   3
// exp(x/8) == exp2(x * 0.125*log2(e)); fold both constants into one multiply
#define SCALE_LOG2E 0.1803368801111204f

// raw v_exp_f32 (logits are tiny; no denormal-range fixup needed)
__device__ __forceinline__ float fast_exp2(float x) {
    return __builtin_amdgcn_exp2f(x);
}

// WGP-scope prefetch: global_prefetch_b8 with default scope (0 = WGP) pulls
// into all cache levels; __builtin_prefetch only reaches SE/DEV/SYS scopes
// which skip the WGP cache.
__device__ __forceinline__ void prefetch_wgp(const void* p) {
    asm volatile("global_prefetch_b8 %0, off" : : "v"(p) : "memory");
}

// workspace layout (bytes); all offsets 16B aligned
#define OFF_QH  (0)
#define OFF_KH  (OFF_QH + B_*HW*C_*2)
#define OFF_VF  (OFF_KH + B_*HW*C_*2)
#define OFF_S   (OFF_VF + B_*HW*C_*4)
#define OFF_WB  (OFF_S  + B_*HW*4)

// --- ISA 7.12.2: 16-bit A (16x32) fragment. lane<16 -> K 0-7 & 16-23,
// lane>=16 -> K 8-15 & 24-31, for row M=lane&15. B (32x16) is symmetric with
// lane = N and K contiguous, which matches row-major (N, K) storage.
__device__ __forceinline__ v16h load_frag16(const _Float16* rowbase, int co) {
    v8h lo = *(const v8h*)(rowbase + co);
    v8h hi = *(const v8h*)(rowbase + co + 16);
    v16h r;
#pragma unroll
    for (int i = 0; i < 8; ++i) { r[i] = lo[i]; r[8 + i] = hi[i]; }
    return r;
}

// ---------------- K1: QKV projection ----------------
__global__ void __launch_bounds__(256)
qkv_proj_kernel(const float* __restrict__ x,
                const float* __restrict__ Wq, const float* __restrict__ bq,
                const float* __restrict__ Wk, const float* __restrict__ bk,
                const float* __restrict__ Wv, const float* __restrict__ bv,
                const float* __restrict__ Wp, const float* __restrict__ bp,
                _Float16* __restrict__ Qh, _Float16* __restrict__ Kh,
                float* __restrict__ Vf)
{
    int tid = blockIdx.x * blockDim.x + threadIdx.x;  // B_*C_*HW threads
    int n = tid & (HW - 1);
    int o = (tid >> 12) & (C_ - 1);
    int b = tid >> 18;
    float q, k, v;
    if (o < QC) {
        q = bq[o]; k = bk[o]; v = bv[o];
        const float* xb = x + (size_t)b * CIN * HW + n;
        for (int c = 0; c < CIN; ++c) {
            float xv = xb[(size_t)c * HW];
            q = fmaf(Wq[o * CIN + c], xv, q);
            k = fmaf(Wk[o * CIN + c], xv, k);
            v = fmaf(Wv[o * CIN + c], xv, v);
        }
    } else {
        int p = o - QC;
        float fi = (float)(n >> 6), fj = (float)(n & 63);
        float pos = fmaf(Wp[p * 2 + 0], fi, fmaf(Wp[p * 2 + 1], fj, bp[p]));
        q = k = v = pos;
    }
    size_t idx = ((size_t)b * HW + n) * C_ + o;
    Qh[idx] = (_Float16)q;
    Kh[idx] = (_Float16)k;
    Vf[idx] = v;
}

// ---------------- K1b: repack Wout -> Wb[kl][o][c] ----------------
__global__ void __launch_bounds__(256)
pack_wout_kernel(const float* __restrict__ Wout, _Float16* __restrict__ Wb)
{
    int tid = blockIdx.x * blockDim.x + threadIdx.x;   // KL*C_*C_ = 200704
    if (tid >= KL * C_ * C_) return;
    int c  = tid & 63;
    int o  = (tid >> 6) & 63;
    int kl = tid >> 12;
    Wb[tid] = (_Float16)Wout[(o * C_ + c) * KL + kl];
}

// ---------------- K2: softmax denominators (full row sums) ----------------
// 4 waves per block; each wave owns one 16-query A tile; all waves share the
// current 16-key B tile staged in LDS (double buffered, software pipelined).
#define RS_WAVES 4
__global__ void __launch_bounds__(32 * RS_WAVES)
rowsum_kernel(const _Float16* __restrict__ Qh, const _Float16* __restrict__ Kh,
              float* __restrict__ S)
{
    __shared__ __align__(16) _Float16 kbuf[2][16 * C_];   // 2 x 2KB key tiles

    int b    = blockIdx.x >> 6;          // grid = B_ * 64
    int tg   = blockIdx.x & 63;          // group of RS_WAVES query tiles
    int wave = threadIdx.x >> 5;
    int lane = threadIdx.x & 31;
    int qt   = tg * RS_WAVES + wave;     // this wave's 16-query tile
    int row  = lane & 15;
    int co   = (lane < 16) ? 0 : 8;

    const _Float16* qrow = Qh + ((size_t)(b * HW) + qt * 16 + row) * C_;
    v16h a0 = load_frag16(qrow, co);        // c = 0..31
    v16h a1 = load_frag16(qrow + 32, co);   // c = 32..63

    float rs[8];
#pragma unroll
    for (int r = 0; r < 8; ++r) rs[r] = 0.0f;

    // cooperative loader: 128 threads cover 16 rows x 8 chunks of 16B
    int lrow = threadIdx.x >> 3;         // 0..15  (key row in tile)
    int lchk = (threadIdx.x & 7) * 8;    // 0,8,...,56 (f16 element offset)
    const _Float16* kbase = Kh + (size_t)(b * HW) * C_;

    // prologue: tile 0 -> buf0
    v8h stage = *(const v8h*)(kbase + (size_t)lrow * C_ + lchk);
    *(v8h*)(&kbuf[0][lrow * C_ + lchk]) = stage;

    for (int mt = 0; mt < HW / 16; ++mt) {
        // issue next tile's global load before compute (hides latency)
        if (mt + 1 < HW / 16)
            stage = *(const v8h*)(kbase + ((size_t)(mt + 1) * 16 + lrow) * C_ + lchk);
        // deep prefetch into the WGP cache (global_prefetch_b8, scope 0)
        if (mt + 8 < HW / 16)
            prefetch_wgp(kbase + ((size_t)(mt + 8) * 16 + lrow) * C_ + lchk);

        __syncthreads();   // kbuf[mt&1] fully written, previous readers done

        const _Float16* kb = &kbuf[mt & 1][row * C_];
        v16h b0 = load_frag16(kb, co);        // ds_load_b128 x2
        v16h b1 = load_frag16(kb + 32, co);
        v8f acc = {};
        acc = __builtin_amdgcn_wmma_f32_16x16x32_f16(false, a0, false, b0,
                                                     (short)0, acc, false, false);
        acc = __builtin_amdgcn_wmma_f32_16x16x32_f16(false, a1, false, b1,
                                                     (short)0, acc, false, false);
        float e[8];
#pragma unroll
        for (int r = 0; r < 8; ++r) e[r] = fast_exp2(acc[r] * SCALE_LOG2E);
        // diag set to 0 before softmax -> exp contribution 1.0 at m == n.
        // C/D layout: lane<16 -> (M=r, N=lane); lane>=16 -> (M=8+r, N=lane-16)
        if (mt == qt) {
            if (lane < 8)   e[lane] = 1.0f;
            if (lane >= 24) e[lane - 24] = 1.0f;
        }
#pragma unroll
        for (int r = 0; r < 8; ++r) rs[r] += e[r];

        // publish next tile into the other buffer
        if (mt + 1 < HW / 16)
            *(v8h*)(&kbuf[(mt + 1) & 1][lrow * C_ + lchk]) = stage;
    }

    // sum across the 16 N-columns held by each lane-half
#pragma unroll
    for (int r = 0; r < 8; ++r) {
        float v = rs[r];
        v += __shfl_xor(v, 1, 32);
        v += __shfl_xor(v, 2, 32);
        v += __shfl_xor(v, 4, 32);
        v += __shfl_xor(v, 8, 32);
        rs[r] = v;
    }
    int qbase = qt * 16;
    if (lane == 0) {
#pragma unroll
        for (int r = 0; r < 8; ++r) S[b * HW + qbase + r] = rs[r];
    } else if (lane == 16) {
#pragma unroll
        for (int r = 0; r < 8; ++r) S[b * HW + qbase + 8 + r] = rs[r];
    }
}

// ---------------- K3: windowed attention * V, conv with Wout ----------------
__global__ void __launch_bounds__(32)
attn_out_kernel(const _Float16* __restrict__ Qh, const _Float16* __restrict__ Kh,
                const float* __restrict__ Vf, const float* __restrict__ S,
                const _Float16* __restrict__ Wb, const float* __restrict__ bout,
                float* __restrict__ out)
{
    __shared__ float s_attn[16][KL];
    __shared__ int   s_m[16][KL];
    __shared__ __align__(16) _Float16 s_a[16][C_];

    int b    = blockIdx.x >> 8;
    int qt   = blockIdx.x & 255;
    int lane = threadIdx.x;
    int qbase = qt * 16;

    // phase 1: 49 clipped-window attention weights per query
    for (int p = lane; p < 16 * KL; p += 32) {
        int q  = p / KL;
        int kl = p % KL;
        int qi = qbase + q;
        int i = qi >> 6, j = qi & 63;
        int ii = i + kl / 7 - R_;  ii = min(max(ii, 0), H_ - 1);
        int jj = j + kl % 7 - R_;  jj = min(max(jj, 0), W_ - 1);
        int m = ii * W_ + jj;
        float num;
        if (m == qi) {
            num = 1.0f;   // diagonal logit forced to 0
        } else {
            const _Float16* qp = Qh + ((size_t)(b * HW) + qi) * C_;
            const _Float16* kp = Kh + ((size_t)(b * HW) + m) * C_;
            float dot = 0.f;
#pragma unroll
            for (int c = 0; c < C_; ++c) dot = fmaf((float)qp[c], (float)kp[c], dot);
            num = fast_exp2(dot * SCALE_LOG2E);   // same formula as denominator
        }
        s_attn[q][kl] = num / S[b * HW + qi];
        s_m[q][kl] = m;
    }
    __syncthreads();

    int row = lane & 15;
    int co  = (lane < 16) ? 0 : 8;
    v8f acc[4];
#pragma unroll
    for (int nt = 0; nt < 4; ++nt) acc[nt] = (v8f){};

    // phase 2: out(16q x 64o) = sum_kl (attn.V)(16q x 64c) @ Wb[kl](64c x 64o)
    for (int kl = 0; kl < KL; ++kl) {
        for (int idx = lane; idx < 16 * C_; idx += 32) {
            int q = idx >> 6, c = idx & 63;
            int m = s_m[q][kl];
            s_a[q][c] = (_Float16)(s_attn[q][kl] * Vf[((size_t)(b * HW) + m) * C_ + c]);
        }
        // prefetch next tap's V rows and Wb tile into the WGP cache
        if (kl + 1 < KL) {
            int mq = s_m[lane & 15][kl + 1];
            prefetch_wgp(Vf + ((size_t)(b * HW) + mq) * C_);
            prefetch_wgp(Wb + ((size_t)(kl + 1) * C_ + lane) * C_);
            prefetch_wgp(Wb + ((size_t)(kl + 1) * C_ + 32 + lane) * C_);
        }
        __syncthreads();
        v16h a0 = load_frag16(&s_a[row][0], co);
        v16h a1 = load_frag16(&s_a[row][0] + 32, co);
        // preload all 8 B fragments so the loads clause/overlap, then 8 WMMAs
        v16h bf0[4], bf1[4];
#pragma unroll
        for (int nt = 0; nt < 4; ++nt) {
            const _Float16* wrow = Wb + ((size_t)kl * C_ + nt * 16 + row) * C_;
            bf0[nt] = load_frag16(wrow, co);
            bf1[nt] = load_frag16(wrow + 32, co);
        }
#pragma unroll
        for (int nt = 0; nt < 4; ++nt) {
            acc[nt] = __builtin_amdgcn_wmma_f32_16x16x32_f16(false, a0, false, bf0[nt],
                                                             (short)0, acc[nt], false, false);
            acc[nt] = __builtin_amdgcn_wmma_f32_16x16x32_f16(false, a1, false, bf1[nt],
                                                             (short)0, acc[nt], false, false);
        }
        __syncthreads();
    }

    // phase 3: bias + relu, store NCHW
#pragma unroll
    for (int nt = 0; nt < 4; ++nt) {
        int o = nt * 16 + row;
        float bo = bout[o];
#pragma unroll
        for (int r = 0; r < 8; ++r) {
            int q = qbase + ((lane < 16) ? r : (8 + r));
            float v = acc[nt][r] + bo;
            out[((size_t)(b * C_) + o) * HW + q] = fmaxf(v, 0.0f);
        }
    }
}

// ---------------------------------------------------------------------------
extern "C" void kernel_launch(void* const* d_in, const int* in_sizes, int n_in,
                              void* d_out, int out_size, void* d_ws, size_t ws_size,
                              hipStream_t stream)
{
    const float* x    = (const float*)d_in[0];
    const float* Wq   = (const float*)d_in[1];
    const float* bq   = (const float*)d_in[2];
    const float* Wk   = (const float*)d_in[3];
    const float* bk   = (const float*)d_in[4];
    const float* Wv   = (const float*)d_in[5];
    const float* bv   = (const float*)d_in[6];
    const float* Wp   = (const float*)d_in[7];
    const float* bp   = (const float*)d_in[8];
    const float* Wout = (const float*)d_in[9];
    const float* bout = (const float*)d_in[10];
    float* out = (float*)d_out;

    char* ws = (char*)d_ws;   // needs ~4.7 MB
    _Float16* Qh = (_Float16*)(ws + OFF_QH);
    _Float16* Kh = (_Float16*)(ws + OFF_KH);
    float*    Vf = (float*)   (ws + OFF_VF);
    float*    Sd = (float*)   (ws + OFF_S);
    _Float16* Wb = (_Float16*)(ws + OFF_WB);

    qkv_proj_kernel<<<dim3((B_ * C_ * HW) / 256), dim3(256), 0, stream>>>(
        x, Wq, bq, Wk, bk, Wv, bv, Wp, bp, Qh, Kh, Vf);
    pack_wout_kernel<<<dim3((KL * C_ * C_ + 255) / 256), dim3(256), 0, stream>>>(Wout, Wb);
    rowsum_kernel<<<dim3(B_ * (HW / 16 / RS_WAVES)), dim3(32 * RS_WAVES), 0, stream>>>(
        Qh, Kh, Sd);
    attn_out_kernel<<<dim3(B_ * (HW / 16)), dim3(32), 0, stream>>>(
        Qh, Kh, Vf, Sd, Wb, bout, out);
}
